// BiQBMVAE_44951127720598
// MI455X (gfx1250) — matrix operand order repaired and checked by reading, not compiled, not run on hardware
//
#include <hip/hip_runtime.h>
#include <hip/hip_bf16.h>
#include <stdint.h>

// ---------- types ----------
typedef __attribute__((ext_vector_type(16))) __bf16         v16bf;
typedef __attribute__((ext_vector_type(8)))  float          v8f;
typedef __attribute__((ext_vector_type(8)))  unsigned short u16x8;
typedef __attribute__((ext_vector_type(16))) unsigned short u16x16;
typedef __attribute__((ext_vector_type(4)))  float          f32x4;
typedef __attribute__((ext_vector_type(4)))  unsigned short u16x4;

__device__ __forceinline__ unsigned short f2bf(float f) {
  unsigned int u = __builtin_bit_cast(unsigned int, f);
  u += 0x7FFFu + ((u >> 16) & 1u);          // round-to-nearest-even
  return (unsigned short)(u >> 16);
}

__device__ __forceinline__ v16bf mkfrag(u16x8 lo, u16x8 hi) {
  u16x16 t;
#pragma unroll
  for (int i = 0; i < 8; ++i) { t[i] = lo[i]; t[i + 8] = hi[i]; }
  return __builtin_bit_cast(v16bf, t);
}

// ---------- elementwise conversion kernels ----------
__global__ __launch_bounds__(256) void sub_convert_bf16(
    const float* __restrict__ x, const float* __restrict__ mean,
    unsigned short* __restrict__ out, int D) {
  const int d = (blockIdx.x * 256 + threadIdx.x) * 4;
  const size_t base = (size_t)blockIdx.y * D + d;
  f32x4 xv = *(const f32x4*)(x + base);
  f32x4 mv = *(const f32x4*)(mean + d);
  u16x4 o;
#pragma unroll
  for (int i = 0; i < 4; ++i) o[i] = f2bf(xv[i] - mv[i]);
  *(u16x4*)(out + base) = o;
}

__global__ __launch_bounds__(256) void cvt_bf16(
    const float* __restrict__ in, unsigned short* __restrict__ out, size_t n) {
  const size_t i = ((size_t)blockIdx.x * 256 + threadIdx.x) * 4;
  if (i >= n) return;
  f32x4 v = *(const f32x4*)(in + i);
  u16x4 o;
#pragma unroll
  for (int j = 0; j < 4; ++j) o[j] = f2bf(v[j]);
  *(u16x4*)(out + i) = o;
}

// ---------- NT GEMM: C[M,N] = A[M,K]·B[N,K]^T, bf16 in / f32 acc ----------
// Block tile 128x128x32, 8 wave32s arranged 4(M)x2(N), wave tile 32x64.
// LDS tiles staged with GLOBAL_LOAD_ASYNC_TO_LDS_B128 (ASYNCcnt), double-buffered.
static constexpr int BM = 128, BN = 128, BK = 32, SK = 40;  // SK: padded LDS row (halves)

// EPI 0: store f32                   (GEMM1 -> h pre-LN)
// EPI 1: +bias, store f32 x3 + bf16  (GEMM2 -> q_logits tuple slots + q bf16)
// EPI 2: sigmoid, store f32          (GEMM3 -> x_recon)
template <int EPI>
__global__ __launch_bounds__(256, 1)
void gemm_nt_bf16(const unsigned short* __restrict__ Amat,
                  const unsigned short* __restrict__ Bmat,
                  int Kdim, int ldc,
                  float* __restrict__ C0, float* __restrict__ C1,
                  float* __restrict__ C2, unsigned short* __restrict__ Cb,
                  const float* __restrict__ bias) {
  __shared__ unsigned short As[2][BM * SK];
  __shared__ unsigned short Bs[2][BN * SK];

  const int t  = threadIdx.x;
  const int m0 = blockIdx.y * BM;
  const int n0 = blockIdx.x * BN;

  // global -> LDS staging: 2 threads per row, 16 halves (32B) each
  const int ldrow = t >> 1;
  const int ldk   = (t & 1) << 4;

  const int lane = t & 31;
  const int wave = t >> 5;
  const int wm   = (wave & 3) * 32;   // wave M offset in block
  const int wn   = (wave >> 2) * 64;  // wave N offset in block
  const int lrow = lane & 15;
  const int lhi  = lane >> 4;         // half-wave select

  v8f acc[2][4];
#pragma unroll
  for (int i = 0; i < 2; ++i)
#pragma unroll
    for (int j = 0; j < 4; ++j) acc[i][j] = v8f{0.f,0.f,0.f,0.f,0.f,0.f,0.f,0.f};

  // Async DMA of this thread's 32B slice of the A and B tiles into LDS.
  // INST_OFFSET is added to BOTH the LDS and global addresses (ISA 15.18.3),
  // so one address pair covers both 16B halves.
  auto load_tile_async = [&](int buf, int kt) {
    const int k0 = kt * BK;
    const unsigned short* ga = Amat + (size_t)(m0 + ldrow) * Kdim + k0 + ldk;
    const unsigned short* gb = Bmat + (size_t)(n0 + ldrow) * Kdim + k0 + ldk;
    const unsigned int la = (unsigned int)(uintptr_t)&As[buf][ldrow * SK + ldk];
    const unsigned int lb = (unsigned int)(uintptr_t)&Bs[buf][ldrow * SK + ldk];
    asm volatile(
        "global_load_async_to_lds_b128 %0, %2, off\n\t"
        "global_load_async_to_lds_b128 %0, %2, off offset:16\n\t"
        "global_load_async_to_lds_b128 %1, %3, off\n\t"
        "global_load_async_to_lds_b128 %1, %3, off offset:16"
        :
        : "v"(la), "v"(lb),
          "v"((unsigned long long)(uintptr_t)ga),
          "v"((unsigned long long)(uintptr_t)gb)
        : "memory");
  };

  const int nk = Kdim / BK;
  load_tile_async(0, 0);
  asm volatile("s_wait_asynccnt 0x0" ::: "memory");
  __syncthreads();

  for (int kt = 0; kt < nk; ++kt) {
    const int cur = kt & 1;
    if (kt + 1 < nk) load_tile_async(cur ^ 1, kt + 1);  // DMA next tile
    if (kt + 2 < nk) {
      const int k2 = (kt + 2) * BK;
      __builtin_prefetch(Amat + (size_t)(m0 + ldrow) * Kdim + k2 + ldk, 0, 0);
      __builtin_prefetch(Bmat + (size_t)(n0 + ldrow) * Kdim + k2 + ldk, 0, 0);
    }

    // A fragments: lane L<16 holds row, K {0..7,16..23}; lane L>=16 K {8..15,24..31}
    v16bf afrag[2], bfrag[4];
#pragma unroll
    for (int mt = 0; mt < 2; ++mt) {
      const int r = wm + mt * 16 + lrow;
      u16x8 lo = *(const u16x8*)&As[cur][r * SK + lhi * 8];
      u16x8 hi = *(const u16x8*)&As[cur][r * SK + lhi * 8 + 16];
      afrag[mt] = mkfrag(lo, hi);
    }
    // B fragments: lane n holds K {0..7,16..23} (lanes 0-15) / {8..15,24..31} (lanes 16-31) of column n
#pragma unroll
    for (int nt = 0; nt < 4; ++nt) {
      const int c = wn + nt * 16 + lrow;
      u16x8 lo = *(const u16x8*)&Bs[cur][c * SK + lhi * 8];
      u16x8 hi = *(const u16x8*)&Bs[cur][c * SK + lhi * 8 + 16];
      bfrag[nt] = mkfrag(lo, hi);
    }

#pragma unroll
    for (int mt = 0; mt < 2; ++mt)
#pragma unroll
      for (int nt = 0; nt < 4; ++nt)
        acc[mt][nt] = __builtin_amdgcn_wmma_f32_16x16x32_bf16(
            false, afrag[mt], false, bfrag[nt], (short)0, acc[mt][nt],
            false, false);

    asm volatile("s_wait_asynccnt 0x0" ::: "memory");  // next tile resident
    __syncthreads();
  }

  // epilogue: C/D layout -> lane(0-15)=N, VGPR r: M=r (lanes<16) / M=r+8 (lanes>=16)
#pragma unroll
  for (int mt = 0; mt < 2; ++mt) {
#pragma unroll
    for (int nt = 0; nt < 4; ++nt) {
      const int mBase = m0 + wm + mt * 16 + lhi * 8;
      const int n     = n0 + wn + nt * 16 + lrow;
#pragma unroll
      for (int r = 0; r < 8; ++r) {
        const size_t idx = (size_t)(mBase + r) * ldc + n;
        float v = acc[mt][nt][r];
        if constexpr (EPI == 0) {
          C0[idx] = v;
        } else if constexpr (EPI == 1) {
          v += bias[n];
          C0[idx] = v; C1[idx] = v; C2[idx] = v;
          Cb[idx] = f2bf(v);
        } else {
          C0[idx] = 1.0f / (1.0f + __expf(-v));
        }
      }
    }
  }
}

// ---------- LayerNorm(+b1) + ReLU -> bf16, one block per row (H=2048) ----------
__global__ __launch_bounds__(256)
void ln_relu_bf16(const float* __restrict__ h, const float* __restrict__ b1,
                  const float* __restrict__ g, const float* __restrict__ bsh,
                  unsigned short* __restrict__ out, int H) {
  __shared__ float s1[256], s2[256];
  const int t = threadIdx.x;
  const size_t base = (size_t)blockIdx.x * H;
  float v[8];
  float sum = 0.f, sq = 0.f;
#pragma unroll
  for (int i = 0; i < 8; ++i) {
    const int c = t + i * 256;
    const float y = h[base + c] + b1[c];
    v[i] = y; sum += y; sq += y * y;
  }
  s1[t] = sum; s2[t] = sq;
  __syncthreads();
  for (int s = 128; s > 0; s >>= 1) {
    if (t < s) { s1[t] += s1[t + s]; s2[t] += s2[t + s]; }
    __syncthreads();
  }
  const float invH = 1.0f / (float)H;
  const float mean = s1[0] * invH;
  const float var  = s2[0] * invH - mean * mean;
  const float rstd = rsqrtf(var + 1e-5f);
#pragma unroll
  for (int i = 0; i < 8; ++i) {
    const int c = t + i * 256;
    float y = (v[i] - mean) * rstd * g[c] + bsh[c];
    y = fmaxf(y, 0.f);
    out[base + c] = f2bf(y);
  }
}

// ---------- launcher ----------
extern "C" void kernel_launch(void* const* d_in, const int* in_sizes, int n_in,
                              void* d_out, int out_size, void* d_ws, size_t ws_size,
                              hipStream_t stream) {
  (void)in_sizes; (void)n_in; (void)out_size; (void)ws_size;
  constexpr int Bb = 8192, D = 4096, H = 2048, K = 512;

  const float* x     = (const float*)d_in[0];
  const float* beta  = (const float*)d_in[1];
  const float* cmean = (const float*)d_in[2];
  const float* W1    = (const float*)d_in[3];
  const float* b1    = (const float*)d_in[4];
  const float* ln_w  = (const float*)d_in[5];
  const float* ln_b  = (const float*)d_in[6];
  const float* W2    = (const float*)d_in[7];
  const float* b2    = (const float*)d_in[8];

  char* ws = (char*)d_ws;
  // layout (bytes); hb/qb alias the xc region (xc dead after GEMM1)
  unsigned short* xc    = (unsigned short*)(ws + 0);          // 67,108,864 B
  unsigned short* hb    = (unsigned short*)(ws + 0);          // 33,554,432 B (alias)
  unsigned short* qb    = (unsigned short*)(ws + 33554432);   //  8,388,608 B (alias)
  unsigned short* w1b   = (unsigned short*)(ws + 67108864);   // 16,777,216 B
  unsigned short* w2b   = (unsigned short*)(ws + 83886080);   //  2,097,152 B
  unsigned short* betab = (unsigned short*)(ws + 85983232);   //  4,194,304 B
  float*          hf    = (float*)(ws + 90177536);            // 67,108,864 B

  float* out = (float*)d_out;
  float* q0 = out;                        // q_logits
  float* xr = out + (size_t)Bb * K;       // x_recon
  float* q1 = xr + (size_t)Bb * D;        // q_logits (copy 2)
  float* q2 = q1 + (size_t)Bb * K;        // q_logits (copy 3)

  // 1) conversions to bf16
  sub_convert_bf16<<<dim3(D / 1024, Bb), 256, 0, stream>>>(x, cmean, xc, D);
  cvt_bf16<<<(size_t)H * D / 1024, 256, 0, stream>>>(W1, w1b, (size_t)H * D);
  cvt_bf16<<<(size_t)K * H / 1024, 256, 0, stream>>>(W2, w2b, (size_t)K * H);
  cvt_bf16<<<(size_t)D * K / 1024, 256, 0, stream>>>(beta, betab, (size_t)D * K);

  // 2) h = xc · W1^T   [8192 x 2048], K=4096
  gemm_nt_bf16<0><<<dim3(H / BN, Bb / BM), 256, 0, stream>>>(
      xc, w1b, D, H, hf, nullptr, nullptr, nullptr, nullptr);

  // 3) LayerNorm(+b1) + ReLU -> bf16
  ln_relu_bf16<<<Bb, 256, 0, stream>>>(hf, b1, ln_w, ln_b, hb, H);

  // 4) q = hb · W2^T + b2   [8192 x 512], K=2048 ; write 3 tuple slots + bf16
  gemm_nt_bf16<1><<<dim3(K / BN, Bb / BM), 256, 0, stream>>>(
      hb, w2b, H, K, q0, q1, q2, qb, b2);

  // 5) x_recon = sigmoid(qb · beta^T)   [8192 x 4096], K=512
  gemm_nt_bf16<2><<<dim3(D / BN, Bb / BM), 256, 0, stream>>>(
      qb, betab, K, D, xr, nullptr, nullptr, nullptr, nullptr);
}
